// model1_85074712199835
// MI455X (gfx1250) — matrix operand order repaired and checked by reading, not compiled, not run on hardware
//
#include <hip/hip_runtime.h>

#define BB 4096
#define TT 256
#define DD 88
#define HH 32
#define KP 96            // K padded to multiple of 32 for bf16 WMMA
#define NEGF (-1e30f)

typedef __attribute__((ext_vector_type(16))) __bf16 v16bf;
typedef __attribute__((ext_vector_type(8)))  float  v8f;

// Workspace layout:
//   [0,   128)   bias   : float[32]   (sum_d log1p(-p))
//   [256, 6400)  W      : bf16[32][96] (log p - log1p(-p), zero padded)
//   [8192, ...)  emit   : float[B*T][32]

__global__ __launch_bounds__(256)
void prep_kernel(const float* __restrict__ py,
                 float* __restrict__ bias,
                 __bf16* __restrict__ w) {
  const int tid = threadIdx.x;
  if (tid < HH) {
    float s = 0.f;
    for (int d = 0; d < DD; ++d) s += log1pf(-py[tid * DD + d]);
    bias[tid] = s;
  }
  for (int idx = tid; idx < HH * KP; idx += blockDim.x) {
    const int h = idx / KP, d = idx % KP;
    float v = 0.f;
    if (d < DD) {
      const float p = py[h * DD + d];
      v = logf(p) - log1pf(-p);
    }
    w[idx] = (__bf16)v;
  }
}

// emit[row, h] = Sum_k seq[row,k] * W[h,k] + bias[h], row = b*T + t
// One wave computes a 16x32 output tile via 6 x v_wmma_f32_16x16x32_bf16.
__global__ __launch_bounds__(256)
void emit_kernel(const float* __restrict__ seq,
                 const float* __restrict__ biasg,
                 const __bf16* __restrict__ wg,
                 float* __restrict__ emit,
                 int numTiles) {
  __shared__ __bf16 sW[HH * KP];
  __shared__ float  sBias[HH];
  for (int i = threadIdx.x; i < HH * KP; i += blockDim.x) sW[i] = wg[i];
  if (threadIdx.x < HH) sBias[threadIdx.x] = biasg[threadIdx.x];
  __syncthreads();

  const int lane       = threadIdx.x & 31;
  const int waveInBlk  = threadIdx.x >> 5;
  const int wavesPerBk = blockDim.x >> 5;
  const int gwave      = blockIdx.x * wavesPerBk + waveInBlk;
  const int nwaves     = gridDim.x * wavesPerBk;

  // B fragments (32x16 bf16 per k-block per n-tile), kept in VGPRs.
  // lane: column n = (lane&15) + 16*nt ; K = 8*(lane>=16) + (j&7) + 16*(j>>3) (table order).
  v16bf bfrag[2][3];
  {
    const int n16   = lane & 15;
    const int khalf = (lane >= 16) ? 8 : 0;
    for (int nt = 0; nt < 2; ++nt) {
      const int n = nt * 16 + n16;
      for (int kb = 0; kb < 3; ++kb) {
        const int kbase = kb * 32 + khalf;
#pragma unroll
        for (int j = 0; j < 16; ++j)
          bfrag[nt][kb][j] = sW[n * KP + kbase + (j & 7) + ((j >> 3) << 4)];
      }
    }
  }

  const int m   = lane & 15;
  const int klo = (lane < 16) ? 0 : 8;   // ISA 16-bit A layout

  for (int tile = gwave; tile < numTiles; tile += nwaves) {
    const int row = tile * 16 + m;
    const float* rowp = seq + (size_t)row * DD;   // 352B row stride, 16B aligned

    // Warm L2 for the next grid-stride tile (gfx1250 global_prefetch path).
    {
      const int ntile = tile + nwaves;
      if (ntile < numTiles) {
        const float* np = seq + (size_t)(ntile * 16 + m) * DD;
        __builtin_prefetch(np, 0, 1);
        __builtin_prefetch(np + 44, 0, 1);   // mid-row: row spans ~3 cachelines
      }
    }

    v8f c0 = {}; v8f c1 = {};
#pragma unroll
    for (int kb = 0; kb < 3; ++kb) {
      v16bf a;
#pragma unroll
      for (int r = 0; r < 2; ++r) {
        const int kbk = kb * 32 + r * 16 + klo;   // multiples of 8; only 88 is OOB
        const bool ok = (kbk <= 80);
        const int  kc = ok ? kbk : 80;            // clamped, always in-bounds
        float4 f0 = *(const float4*)(rowp + kc);
        float4 f1 = *(const float4*)(rowp + kc + 4);
        const float z = ok ? 1.f : 0.f;           // zero the padded K run
        a[r*8+0] = (__bf16)(f0.x * z); a[r*8+1] = (__bf16)(f0.y * z);
        a[r*8+2] = (__bf16)(f0.z * z); a[r*8+3] = (__bf16)(f0.w * z);
        a[r*8+4] = (__bf16)(f1.x * z); a[r*8+5] = (__bf16)(f1.y * z);
        a[r*8+6] = (__bf16)(f1.z * z); a[r*8+7] = (__bf16)(f1.w * z);
      }
      c0 = __builtin_amdgcn_wmma_f32_16x16x32_bf16(false, a, false, bfrag[0][kb],
                                                   (short)0, c0, false, false);
      c1 = __builtin_amdgcn_wmma_f32_16x16x32_bf16(false, a, false, bfrag[1][kb],
                                                   (short)0, c1, false, false);
    }

    // C layout: VGPR r -> M = r (+8 for lanes 16..31); N = lane&15
    const int mbase = tile * 16 + ((lane >> 4) << 3);
    const int n16   = lane & 15;
    const float b0 = sBias[n16], b1 = sBias[16 + n16];
#pragma unroll
    for (int r = 0; r < 8; ++r) {
      const size_t orow = (size_t)(mbase + r) * HH;
      emit[orow + n16]      = c0[r] + b0;
      emit[orow + 16 + n16] = c1[r] + b1;
    }
  }
}

// Forward recursion: one wave per sequence, lane = hidden state.
// logsumexp step done in linear space: exp-shift, 32-FMA matvec, log.
// emit loads are independent of the recursion -> one-step software prefetch.
__global__ __launch_bounds__(256)
void hmm_fwd_kernel(const float* __restrict__ emit,
                    const float* __restrict__ px,
                    const int* __restrict__ lengths,
                    float* __restrict__ out) {
  const int lane = threadIdx.x & 31;
  const int b = (blockIdx.x * blockDim.x + threadIdx.x) >> 5;
  if (b >= BB) return;

  // Column 'lane' of the transition matrix, register resident.
  float tc[HH];
#pragma unroll
  for (int j = 0; j < HH; ++j) tc[j] = px[j * HH + lane];

  const int len = lengths[b];
  float alpha = (lane == 0) ? 0.f : NEGF;
  const float* eb = emit + (size_t)b * TT * HH;

  float e = eb[lane];                      // t = 0 row
  for (int t = 0; t < TT; ++t) {
    // issue next row's load before the dependent exp/shuffle/log chain
    const int tn = (t + 1 < TT) ? (t + 1) : t;
    float e_next = eb[tn * HH + lane];

    float mx = alpha;
#pragma unroll
    for (int off = 16; off > 0; off >>= 1)
      mx = fmaxf(mx, __shfl_xor(mx, off, 32));
    const float p = __expf(alpha - mx);
    float acc = 0.f;
#pragma unroll
    for (int h = 0; h < HH; ++h)
      acc = fmaf(__shfl(p, h, 32), tc[h], acc);
    const float na = __logf(acc) + mx + e;
    alpha = (t < len) ? na : alpha;

    e = e_next;
  }

  float mx = alpha;
#pragma unroll
  for (int off = 16; off > 0; off >>= 1)
    mx = fmaxf(mx, __shfl_xor(mx, off, 32));
  float s = __expf(alpha - mx);
#pragma unroll
  for (int off = 16; off > 0; off >>= 1)
    s += __shfl_xor(s, off, 32);
  if (lane == 0) out[b] = __logf(s) + mx;
}

extern "C" void kernel_launch(void* const* d_in, const int* in_sizes, int n_in,
                              void* d_out, int out_size, void* d_ws, size_t ws_size,
                              hipStream_t stream) {
  const float* seq     = (const float*)d_in[0];   // [B,T,D] f32 (binary)
  const int*   lengths = (const int*)  d_in[1];   // [B]
  const float* px      = (const float*)d_in[2];   // [H,H]
  const float* py      = (const float*)d_in[3];   // [H,D]
  float* out = (float*)d_out;

  char* ws = (char*)d_ws;
  float*  bias = (float*)ws;
  __bf16* w    = (__bf16*)(ws + 256);
  float*  emit = (float*)(ws + 8192);   // B*T*H f32 = 128 MiB

  prep_kernel<<<1, 256, 0, stream>>>(py, bias, w);

  const int numTiles = (BB * TT) / 16;  // 65536 M-tiles
  emit_kernel<<<2048, 256, 0, stream>>>(seq, bias, w, emit, numTiles);

  hmm_fwd_kernel<<<BB / 8, 256, 0, stream>>>(emit, px, lengths, out);
}
